// MambaBlock_23493471109479
// MI455X (gfx1250) — matrix operand order repaired and checked by reading, not compiled, not run on hardware
//
#include <hip/hip_runtime.h>
#include <hip/hip_bf16.h>
#include <math.h>

// ---------------------------------------------------------------------------
// Mamba block for MI455X (gfx1250, wave32, WMMA).
// All GEMMs: bf16 operands pre-converted in memory (one conversion per
// element, not per consuming wave), V_WMMA_F32_16X16X32_BF16 with fp32
// accumulate. Hot loop is pure global_load_b128 + wmma: zero conversion VALU,
// zero predication (x_proj is padded from N=160 to N=256 so every launch is
// tile-aligned).
// ---------------------------------------------------------------------------

typedef __attribute__((ext_vector_type(16))) __bf16 v16bf;
typedef __attribute__((ext_vector_type(8)))  __bf16 v8bf;
typedef __attribute__((ext_vector_type(8)))  float  v8f;

#define B_      2
#define L_      1024
#define DMODEL  2048
#define DINNER  4096
#define DSTATE  16
#define DCONV   4
#define DTRANK  128
#define XDIM    (DTRANK + 2 * DSTATE)   // 160 (logical)
#define XPAD    256                      // padded leading dim / padded N
#define MROWS   (B_ * L_)                // 2048
#define EPSV    1e-5f

// ---- fp32 -> bf16 (RNE) and bf16 -> fp32 ---------------------------------
static __device__ inline __bf16 f2bf(float f) {
    unsigned u = __builtin_bit_cast(unsigned, f);
    u += 0x7FFFu + ((u >> 16) & 1u);
    unsigned short h = (unsigned short)(u >> 16);
    return __builtin_bit_cast(__bf16, h);
}
static __device__ inline float bf2f(__bf16 b) {
    unsigned short h = __builtin_bit_cast(unsigned short, b);
    unsigned u = ((unsigned)h) << 16;
    return __builtin_bit_cast(float, u);
}

// ---- A fragment: 16x32 bf16 from bf16 memory, row-major, lda elements.
//      Lane l: row (l&15); K-chunks [kh*8, kh*8+8) and [16+kh*8, ...). ------
static __device__ inline v16bf load_a_frag(const __bf16* __restrict__ A, int lda,
                                           int row0, int k0, int lane) {
    const int m  = row0 + (lane & 15);
    const int kh = (lane >> 4) & 1;
    const __bf16* p = A + (size_t)m * lda + k0 + kh * 8;
    v8bf lo = *(const v8bf*)p;          // one global_load_b128
    v8bf hi = *(const v8bf*)(p + 16);   // one global_load_b128
    v16bf r;
#pragma unroll
    for (int i = 0; i < 8; ++i) { r[i] = lo[i]; r[8 + i] = hi[i]; }
    return r;
}

// ---- B fragment: 32x16 bf16 from W[N,K] row-major bf16 (transposed weight).
//      Lane l: column (l&15); K-chunks [kh*8, kh*8+8) and [16+kh*8, ...). ------
static __device__ inline v16bf load_b_frag(const __bf16* __restrict__ W, int ldb,
                                           int col0, int k0, int lane) {
    const int n  = col0 + (lane & 15);
    const int ks = k0 + ((lane >> 4) & 1) * 8;
    const __bf16* p = W + (size_t)n * ldb + ks;
    v8bf lo = *(const v8bf*)p;
    v8bf hi = *(const v8bf*)(p + 16);
    v16bf r;
#pragma unroll
    for (int i = 0; i < 8; ++i) { r[i] = lo[i]; r[8 + i] = hi[i]; }
    return r;
}

// ---- Tiled WMMA GEMM:  C[M,N] = A[M,K] * W[N,K]^T  -------------------------
// Block tile 128x128, 8 waves, wave tile 32x64 (2x4 WMMA tiles), K-step 32.
// M, N always multiples of 128 (by construction) -> no predication.
// EPI 0: fp32 store. EPI 1: softplus(x + bias[n]) fp32 store (dt_proj).
// EPI 2: fp32 store + bf16 mirror store to C2 (x_proj feeding dt_proj).
template <int EPI>
__global__ __launch_bounds__(256) void gemm_wmma_bf16(
    const __bf16* __restrict__ A, int lda,
    const __bf16* __restrict__ W, int ldb,
    float* __restrict__ C, __bf16* __restrict__ C2, int ldc,
    int K, const float* __restrict__ bias)
{
    const int lane = threadIdx.x & 31;
    const int wave = threadIdx.x >> 5;
    const int row0 = blockIdx.y * 128 + (wave & 3) * 32;   // 4 waves along M
    const int col0 = blockIdx.x * 128 + (wave >> 2) * 64;  // 2 waves along N

    v8f acc[2][4];
#pragma unroll
    for (int i = 0; i < 2; ++i)
#pragma unroll
        for (int j = 0; j < 4; ++j)
#pragma unroll
            for (int e = 0; e < 8; ++e) acc[i][j][e] = 0.f;

    for (int k0 = 0; k0 < K; k0 += 32) {
        v16bf a0 = load_a_frag(A, lda, row0,      k0, lane);
        v16bf a1 = load_a_frag(A, lda, row0 + 16, k0, lane);
        v16bf b0 = load_b_frag(W, ldb, col0,      k0, lane);
        v16bf b1 = load_b_frag(W, ldb, col0 + 16, k0, lane);
        v16bf b2 = load_b_frag(W, ldb, col0 + 32, k0, lane);
        v16bf b3 = load_b_frag(W, ldb, col0 + 48, k0, lane);

        acc[0][0] = __builtin_amdgcn_wmma_f32_16x16x32_bf16(false, a0, false, b0, (short)0, acc[0][0], false, false);
        acc[0][1] = __builtin_amdgcn_wmma_f32_16x16x32_bf16(false, a0, false, b1, (short)0, acc[0][1], false, false);
        acc[0][2] = __builtin_amdgcn_wmma_f32_16x16x32_bf16(false, a0, false, b2, (short)0, acc[0][2], false, false);
        acc[0][3] = __builtin_amdgcn_wmma_f32_16x16x32_bf16(false, a0, false, b3, (short)0, acc[0][3], false, false);
        acc[1][0] = __builtin_amdgcn_wmma_f32_16x16x32_bf16(false, a1, false, b0, (short)0, acc[1][0], false, false);
        acc[1][1] = __builtin_amdgcn_wmma_f32_16x16x32_bf16(false, a1, false, b1, (short)0, acc[1][1], false, false);
        acc[1][2] = __builtin_amdgcn_wmma_f32_16x16x32_bf16(false, a1, false, b2, (short)0, acc[1][2], false, false);
        acc[1][3] = __builtin_amdgcn_wmma_f32_16x16x32_bf16(false, a1, false, b3, (short)0, acc[1][3], false, false);
    }

    // C/D layout: VGPR r -> row (r + 8*kh), lane -> column (lane & 15)
    const int nlo = lane & 15;
    const int kh  = (lane >> 4) & 1;
#pragma unroll
    for (int i = 0; i < 2; ++i) {
#pragma unroll
        for (int j = 0; j < 4; ++j) {
            const int n = col0 + 16 * j + nlo;
            const float bv = (EPI == 1) ? bias[n] : 0.f;
#pragma unroll
            for (int r8 = 0; r8 < 8; ++r8) {
                const int m = row0 + 16 * i + 8 * kh + r8;
                float v = acc[i][j][r8];
                if (EPI == 1) {  // softplus(v + bias)
                    v += bv;
                    v = (v > 20.f) ? v : log1pf(__expf(v));
                }
                C[(size_t)m * ldc + n] = v;
                if (EPI == 2) C2[(size_t)m * ldc + n] = f2bf(v);
            }
        }
    }
}

// ---- weight fp32 -> bf16, 4 elements per thread (count % 1024 == 0) -------
__global__ __launch_bounds__(256) void cvt_w_kernel(
    const float* __restrict__ src, __bf16* __restrict__ dst)
{
    const size_t i4 = ((size_t)blockIdx.x * 256 + threadIdx.x) * 4;
    const float4 v = *(const float4*)(src + i4);
    dst[i4 + 0] = f2bf(v.x);
    dst[i4 + 1] = f2bf(v.y);
    dst[i4 + 2] = f2bf(v.z);
    dst[i4 + 3] = f2bf(v.w);
}

// ---- x_proj weight: convert [160,4096] and zero-pad rows to [256,4096] ----
__global__ __launch_bounds__(256) void cvt_w_pad_kernel(
    const float* __restrict__ src, __bf16* __restrict__ dst)
{
    const size_t idx = (size_t)blockIdx.x * 256 + threadIdx.x;  // < 256*4096
    const int n = (int)(idx / DINNER);
    dst[idx] = (n < XDIM) ? f2bf(src[idx - (size_t)(n - 0) * 0 - 0]) : f2bf(0.f);
}

// (note: for n < XDIM the padded and source indices coincide: idx = n*DINNER+k)

// ---- fused residual add + RMSNorm; fp32 residual out + bf16 normed out ----
__global__ __launch_bounds__(256) void addnorm_kernel(
    const float* __restrict__ h, const float* __restrict__ r,
    const float* __restrict__ w, float* __restrict__ resid_out,
    __bf16* __restrict__ hs_bf)
{
    const int row = blockIdx.x;          // b*L + l
    const int tid = threadIdx.x;
    const size_t base = (size_t)row * DMODEL;
    __shared__ float red[256];

    float ss = 0.f;
    for (int i = tid; i < DMODEL; i += 256) {
        const float v = h[base + i] + r[base + i];
        resid_out[base + i] = v;
        ss += v * v;
    }
    red[tid] = ss;
    __syncthreads();
    for (int s = 128; s > 0; s >>= 1) {
        if (tid < s) red[tid] += red[tid + s];
        __syncthreads();
    }
    const float inv = rsqrtf(red[0] / (float)DMODEL + EPSV);
    for (int i = tid; i < DMODEL; i += 256) {
        hs_bf[base + i] = f2bf(resid_out[base + i] * inv * w[i]);
    }
}

// ---- depthwise causal conv (d_conv=4) + SiLU; bf16 in, f32+bf16 out -------
__global__ __launch_bounds__(256) void conv_silu_kernel(
    const __bf16* __restrict__ xz, const float* __restrict__ cw,
    const float* __restrict__ cb, float* __restrict__ xconv_f,
    __bf16* __restrict__ xconv_bf)
{
    const size_t idx = (size_t)blockIdx.x * 256 + threadIdx.x;  // < MROWS*DINNER
    const int d   = (int)(idx % DINNER);
    const int row = (int)(idx / DINNER);   // b*L + l
    const int l   = row % L_;

    float acc = cb[d];
#pragma unroll
    for (int j = 0; j < DCONV; ++j) {
        const int ls = l - (DCONV - 1) + j;
        if (ls >= 0)
            acc += bf2f(xz[(size_t)(row - (DCONV - 1) + j) * (2 * DINNER) + d]) * cw[d * DCONV + j];
    }
    const float sig = 1.f / (1.f + __expf(-acc));
    const float out = acc * sig;
    xconv_f[idx]  = out;
    xconv_bf[idx] = f2bf(out);
}

// ---- selective scan: one thread per (b, d) channel, 16 states in regs -----
__global__ __launch_bounds__(256) void scan_kernel(
    const float* __restrict__ dtb, const float* __restrict__ xconv,
    const float* __restrict__ xdbl, const float* __restrict__ A_log,
    const float* __restrict__ Dp, float* __restrict__ y)
{
    const int d = (blockIdx.x & 15) * 256 + threadIdx.x;  // 0..4095
    const int b = blockIdx.x >> 4;                        // 0..1

    float Av[DSTATE], h[DSTATE];
#pragma unroll
    for (int s = 0; s < DSTATE; ++s) {
        Av[s] = -__expf(A_log[d * DSTATE + s]);
        h[s]  = 0.f;
    }
    const float Dv = Dp[d];

    for (int l = 0; l < L_; ++l) {
        const size_t row = (size_t)b * L_ + l;
        const float dtv = dtb[row * DINNER + d];
        const float xv  = xconv[row * DINNER + d];
        const float dx  = dtv * xv;
        const float* bc = xdbl + row * XPAD;   // wave-uniform -> L2 broadcast
        float acc = 0.f;
#pragma unroll
        for (int s = 0; s < DSTATE; ++s) {
            const float Bv = bc[DTRANK + s];
            const float Cv = bc[DTRANK + DSTATE + s];
            h[s] = __expf(dtv * Av[s]) * h[s] + dx * Bv;
            acc += h[s] * Cv;
        }
        y[row * DINNER + d] = acc + xv * Dv;
    }
}

// ---- yg = bf16( y * silu(z) ) ---------------------------------------------
__global__ __launch_bounds__(256) void gate_kernel(
    const __bf16* __restrict__ xz, const float* __restrict__ y,
    __bf16* __restrict__ yg)
{
    const size_t idx = (size_t)blockIdx.x * 256 + threadIdx.x;
    const int d     = (int)(idx % DINNER);
    const size_t rw = idx / DINNER;
    const float z   = bf2f(xz[rw * (2 * DINNER) + DINNER + d]);
    const float sz  = z / (1.f + __expf(-z));
    yg[idx] = f2bf(y[idx] * sz);
}

// ---------------------------------------------------------------------------
static inline size_t align256(size_t x) { return (x + 255) & ~(size_t)255; }

extern "C" void kernel_launch(void* const* d_in, const int* in_sizes, int n_in,
                              void* d_out, int out_size, void* d_ws, size_t ws_size,
                              hipStream_t stream)
{
    (void)in_sizes; (void)n_in; (void)out_size; (void)ws_size;

    const float* hidden     = (const float*)d_in[0];
    const float* resid_in   = (const float*)d_in[1];
    const float* norm_w     = (const float*)d_in[2];
    const float* in_proj_w  = (const float*)d_in[3];   // [8192, 2048]
    const float* conv_w     = (const float*)d_in[4];   // [4096, 4]
    const float* conv_b     = (const float*)d_in[5];
    const float* x_proj_w   = (const float*)d_in[6];   // [160, 4096]
    const float* dt_proj_w  = (const float*)d_in[7];   // [4096, 128]
    const float* dt_proj_b  = (const float*)d_in[8];
    const float* A_log      = (const float*)d_in[9];   // [4096, 16]
    const float* D_param    = (const float*)d_in[10];
    const float* out_proj_w = (const float*)d_in[11];  // [2048, 4096]

    float* out       = (float*)d_out;                       // [MROWS, DMODEL]
    float* resid_out = out + (size_t)MROWS * DMODEL;        // [MROWS, DMODEL]

    // ---- workspace carve (all 256B aligned) ----
    char* p = (char*)d_ws;
    auto carve = [&](size_t bytes) -> void* { void* r = (void*)p; p += align256(bytes); return r; };

    __bf16* hs_bf    = (__bf16*)carve((size_t)MROWS * DMODEL * 2);        // normed activations
    __bf16* xz_bf    = (__bf16*)carve((size_t)MROWS * 2 * DINNER * 2);    // in_proj output
    float*  xconv_f  = (float*) carve((size_t)MROWS * DINNER * 4);
    __bf16* xconv_bf = (__bf16*)carve((size_t)MROWS * DINNER * 2);
    float*  xdbl_f   = (float*) carve((size_t)MROWS * XPAD * 4);          // padded ldc=256
    __bf16* xdbl_bf  = (__bf16*)carve((size_t)MROWS * XPAD * 2);
    float*  dtb_f    = (float*) carve((size_t)MROWS * DINNER * 4);
    float*  yb_f     = (float*) carve((size_t)MROWS * DINNER * 4);
    __bf16* yg_bf    = (__bf16*)carve((size_t)MROWS * DINNER * 2);
    __bf16* w_in_bf  = (__bf16*)carve((size_t)2 * DINNER * DMODEL * 2);
    __bf16* w_xp_bf  = (__bf16*)carve((size_t)XPAD * DINNER * 2);         // zero-padded rows
    __bf16* w_dt_bf  = (__bf16*)carve((size_t)DINNER * DTRANK * 2);
    __bf16* w_out_bf = (__bf16*)carve((size_t)DMODEL * DINNER * 2);

    // ---- one-time weight conversions (each element converted exactly once) ----
    cvt_w_kernel<<<(2 * DINNER * DMODEL) / 1024, 256, 0, stream>>>(in_proj_w,  w_in_bf);
    cvt_w_pad_kernel<<<(XPAD * DINNER) / 256, 256, 0, stream>>>(x_proj_w, w_xp_bf);
    cvt_w_kernel<<<(DINNER * DTRANK) / 1024, 256, 0, stream>>>(dt_proj_w,  w_dt_bf);
    cvt_w_kernel<<<(DMODEL * DINNER) / 1024, 256, 0, stream>>>(out_proj_w, w_out_bf);

    // 1) residual = h + r ; hs = bf16(rmsnorm(residual) * w)
    addnorm_kernel<<<MROWS, 256, 0, stream>>>(hidden, resid_in, norm_w, resid_out, hs_bf);

    // 2) xz = hs @ in_proj_w^T   [2048 x 8192] -> bf16 via EPI2 with C=f32 scratch?
    //    z-half is only consumed through silu (bf16 OK) and x-half through conv
    //    (bf16 OK), so store bf16 directly: reuse EPI2 with fp32 sink = xdbl? No:
    //    simplest correct path = EPI2 writing both, but we only need bf16 here.
    //    We use EPI2 with a dedicated f32 sink reused later? Cleaner: EPI2 writes
    //    fp32 to yb_f-sized region is wasteful; instead EPI0-f32 would force conv
    //    to read fp32. We accept one fp32 sink: write fp32 into dtb_f (dead until
    //    step 5) and bf16 into xz_bf. dtb_f is fully overwritten at step 5.
    gemm_wmma_bf16<2><<<dim3((2 * DINNER) / 128, MROWS / 128), 256, 0, stream>>>(
        hs_bf, DMODEL, w_in_bf, DMODEL, /*C f32 sink (first half) =*/ (float*)dtb_f, xz_bf,
        2 * DINNER, DMODEL, nullptr);

    // 3) depthwise causal conv + SiLU on x half of xz (bf16 in, f32+bf16 out)
    conv_silu_kernel<<<(MROWS * DINNER) / 256, 256, 0, stream>>>(
        xz_bf, conv_w, conv_b, xconv_f, xconv_bf);

    // 4) x_dbl = xconv @ x_proj_w^T   [2048 x 256(padded)] -> f32 + bf16
    gemm_wmma_bf16<2><<<dim3(XPAD / 128, MROWS / 128), 256, 0, stream>>>(
        xconv_bf, DINNER, w_xp_bf, DINNER, xdbl_f, xdbl_bf, XPAD, DINNER, nullptr);

    // 5) dt = softplus(x_dbl[:, :128] @ dt_proj_w^T + b)   [2048 x 4096] f32
    gemm_wmma_bf16<1><<<dim3(DINNER / 128, MROWS / 128), 256, 0, stream>>>(
        xdbl_bf, XPAD, w_dt_bf, DTRANK, dtb_f, nullptr, DINNER, DTRANK, dt_proj_b);

    // 6) selective scan -> y f32
    scan_kernel<<<B_ * (DINNER / 256), 256, 0, stream>>>(
        dtb_f, xconv_f, xdbl_f, A_log, D_param, yb_f);

    // 7) yg = bf16( y * silu(z) )
    gate_kernel<<<(MROWS * DINNER) / 256, 256, 0, stream>>>(xz_bf, yb_f, yg_bf);

    // 8) out = yg @ out_proj_w^T   [2048 x 2048] f32 -> d_out
    gemm_wmma_bf16<0><<<dim3(DMODEL / 128, MROWS / 128), 256, 0, stream>>>(
        yg_bf, DINNER, w_out_bf, DINNER, out, nullptr, DMODEL, DINNER, nullptr);
}
